// GTCRN_4020089389527
// MI455X (gfx1250) — matrix operand order repaired and checked by reading, not compiled, not run on hardware
//
#include <hip/hip_runtime.h>
#include <hip/hip_bf16.h>

typedef __attribute__((ext_vector_type(16))) _Float16 v16h;
typedef __attribute__((ext_vector_type(8)))  _Float16 v8h;
typedef __attribute__((ext_vector_type(8)))  float    v8f;

union AFrag { v16h v; v8h h[2]; };

#define TT 512
#define FF 256
#define BB 8
#define HT ((size_t)BB * 16 * TT * FF)   // elements in one (B,16,T,F) tensor

#define LOG2E 1.44269504088896340736f

// Fast transcendentals on the CDNA5 TRANS pipe: v_exp_f32 / v_rcp_f32.
__device__ __forceinline__ float fexp2(float x) { return __builtin_amdgcn_exp2f(x); }
__device__ __forceinline__ float sigf(float x) {
    // 1/(1+e^-x); e^-x -> +inf gives rcp(inf)=0, -> 0 gives 1: saturates cleanly.
    return __builtin_amdgcn_rcpf(1.0f + fexp2(-x * LOG2E));
}
__device__ __forceinline__ float tanh_fast(float x) {
    // 1 - 2/(e^{2x}+1); saturates to +/-1 without inf/inf NaN.
    return 1.0f - 2.0f * __builtin_amdgcn_rcpf(1.0f + fexp2(2.0f * LOG2E * x));
}

// ---------------------------------------------------------------------------
// Prep (gates 2-4, Cin=16): bake conv weights into WMMA B-fragment layout
// (f16) and fold bias+BN into per-channel scale/shift.
// wfrag[(q*32+lane)*16+j], q = nt*5+i : lane n<16 -> K=j (tap=2i),
// n>=16 -> K=16+j (tap=2i+1), cin = j, oc = nt*16 + (n&15).
// ---------------------------------------------------------------------------
__global__ void prep_kernel(const float* __restrict__ w, const float* __restrict__ bias,
                            const float* __restrict__ gamma, const float* __restrict__ beta,
                            const float* __restrict__ mean, const float* __restrict__ var,
                            int Cin, _Float16* __restrict__ wfrag, float* __restrict__ scsh) {
    int tid = threadIdx.x;                 // 320 threads
    int q = tid >> 5, lane = tid & 31;
    int nt = q / 5, i = q % 5;
    int tapbit = (lane >= 16) ? 1 : 0;
    int tap = 2 * i + tapbit;
    int oc = nt * 16 + (lane & 15);
    _Float16* out = wfrag + ((size_t)(q * 32 + lane) << 4);
    for (int j = 0; j < 16; ++j) {
        float v = 0.0f;
        if (tap < 9 && j < Cin) {
            int kt = tap / 3, kf = tap % 3;
            v = w[((oc * Cin + j) * 3 + kt) * 3 + kf];
        }
        out[j] = (_Float16)v;
    }
    if (tid < 32) {
        scsh[tid] = gamma[tid] * rsqrtf(var[tid] + 1e-5f);
    } else if (tid < 64) {
        int oc2 = tid - 32;
        float sc = gamma[oc2] * rsqrtf(var[oc2] + 1e-5f);
        scsh[32 + oc2] = (bias[oc2] - mean[oc2]) * sc + beta[oc2];
    }
}

// ---------------------------------------------------------------------------
// Prep for gate1 (Cin=2): single K=32 step, k = tap*2 + cin (taps 9..15 zero).
// wfrag[(nt*32+lane)*16+j]: lane n<16 -> K=j, n>=16 -> K=16+j.
// ---------------------------------------------------------------------------
__global__ void prep1_kernel(const float* __restrict__ w, const float* __restrict__ bias,
                             const float* __restrict__ gamma, const float* __restrict__ beta,
                             const float* __restrict__ mean, const float* __restrict__ var,
                             _Float16* __restrict__ wfrag, float* __restrict__ scsh) {
    int tid = threadIdx.x;                 // 64 threads
    int nt = tid >> 5, lane = tid & 31;
    int oc = nt * 16 + (lane & 15);
    int kbase = (lane >= 16) ? 16 : 0;
    _Float16* out = wfrag + ((size_t)(nt * 32 + lane) << 4);
    for (int j = 0; j < 16; ++j) {
        int k = kbase + j;
        int tap = k >> 1, cin = k & 1;
        float v = 0.0f;
        if (tap < 9)
            v = w[((oc * 2 + cin) * 3 + tap / 3) * 3 + tap % 3];
        out[j] = (_Float16)v;
    }
    if (tid < 32) {
        scsh[tid] = gamma[tid] * rsqrtf(var[tid] + 1e-5f);
    } else {
        int oc2 = tid - 32;
        float sc = gamma[oc2] * rsqrtf(var[oc2] + 1e-5f);
        scsh[32 + oc2] = (bias[oc2] - mean[oc2]) * sc + beta[oc2];
    }
}

// ---------------------------------------------------------------------------
// Gate1: Cin=2, dil=1, single K=32 WMMA step per tile.
// LDS holds A-lines in K order per output position: ldsA[f][k], k = tap*2+cin.
// ---------------------------------------------------------------------------
__global__ void gate_conv1(const float* __restrict__ src, const _Float16* __restrict__ wfrag,
                           const float* __restrict__ scsh, _Float16* __restrict__ dst) {
    const int t = blockIdx.x, b = blockIdx.y;
    __shared__ _Float16 ldsA[256 * 32];   // 16 KB

    for (int idx = threadIdx.x; idx < 256 * 32; idx += blockDim.x) {
        int k = idx >> 8, f = idx & 255;
        int tap = k >> 1, cin = k & 1;
        float v = 0.0f;
        if (tap < 9) {
            int row = t + tap / 3 - 1;
            int fs = f + tap % 3 - 1;
            if (row >= 0 && row < TT && (unsigned)fs < (unsigned)FF)
                v = src[(((size_t)b * 2 + cin) * TT + row) * FF + fs];
        }
        ldsA[f * 32 + k] = (_Float16)v;
    }
    __syncthreads();

    const int wave = threadIdx.x >> 5, lane = threadIdx.x & 31;
    const int M = lane & 15;
    const int hi = lane >> 4;

    AFrag bf0, bf1;
    {
        const _Float16* p0 = wfrag + ((size_t)(0 * 32 + lane) << 4);
        const _Float16* p1 = wfrag + ((size_t)(1 * 32 + lane) << 4);
        bf0.h[0] = *(const v8h*)p0;  bf0.h[1] = *(const v8h*)(p0 + 8);
        bf1.h[0] = *(const v8h*)p1;  bf1.h[1] = *(const v8h*)(p1 + 8);
    }

    const v8f vzero = {0.f, 0.f, 0.f, 0.f, 0.f, 0.f, 0.f, 0.f};
    v8f acc[2][2];
    acc[0][0] = vzero; acc[0][1] = vzero; acc[1][0] = vzero; acc[1][1] = vzero;

    #pragma unroll
    for (int ft = 0; ft < 2; ++ft) {
        const int f0 = (wave * 2 + ft) * 16;
        const _Float16* ap = &ldsA[(f0 + M) * 32 + hi * 16];
        AFrag af;
        af.h[0] = *(const v8h*)ap;          // K = hi*16 .. hi*16+7
        af.h[1] = *(const v8h*)(ap + 8);    // K = hi*16+8 .. hi*16+15
        acc[ft][0] = __builtin_amdgcn_wmma_f32_16x16x32_f16(
            false, af.v, false, bf0.v, (short)0, acc[ft][0], false, false);
        acc[ft][1] = __builtin_amdgcn_wmma_f32_16x16x32_f16(
            false, af.v, false, bf1.v, (short)0, acc[ft][1], false, false);
    }

    const int ch = lane & 15;
    const float sca = scsh[ch],      scg = scsh[16 + ch];
    const float sha = scsh[32 + ch], shg = scsh[48 + ch];
    for (int ft = 0; ft < 2; ++ft) {
        const int f0 = (wave * 2 + ft) * 16 + hi * 8;
        v8h outv;
        #pragma unroll
        for (int r = 0; r < 8; ++r) {
            float a = acc[ft][0][r] * sca + sha;
            float g = acc[ft][1][r] * scg + shg;
            outv[r] = (_Float16)(tanh_fast(a) * sigf(g));
        }
        *(v8h*)&dst[(((size_t)b * 16 + ch) * TT + t) * FF + f0] = outv;
    }
}

// ---------------------------------------------------------------------------
// Gates 2-4: Cin=16 dilated 3x3 conv + BN + tanh*sigmoid, implicit GEMM WMMA.
// One workgroup per (t,b) row. 8 waves; wave handles 2 f-tiles of 16.
// K packed: 5 WMMA steps of K=32 = taps (2i, 2i+1) x 16 cin slots.
// ---------------------------------------------------------------------------
__global__ void gate_conv(const _Float16* __restrict__ src, const _Float16* __restrict__ wfrag,
                          const float* __restrict__ scsh, _Float16* __restrict__ dst, int dil) {
    const int t = blockIdx.x, b = blockIdx.y;
    const int W = FF + 2 * dil;
    __shared__ _Float16 xlds[3 * 272 * 16];   // [row][fpos][cin] halves, max dil=8

    __builtin_prefetch(wfrag, 0, 1);

    const int total = 3 * 16 * W;
    for (int k = threadIdx.x; k < total; k += blockDim.x) {
        int dt = k / (16 * W);
        int rcm = k - dt * 16 * W;
        int cin = rcm / W;
        int j = rcm % W;                  // consecutive lanes -> consecutive f
        int row = t + (dt - 1) * dil;
        int fs = j - dil;
        float v = 0.0f;
        if (row >= 0 && row < TT && fs >= 0 && fs < FF)
            v = (float)src[(((size_t)b * 16 + cin) * TT + row) * FF + fs];
        xlds[(dt * W + j) * 16 + cin] = (_Float16)v;
    }
    __syncthreads();

    const int wave = threadIdx.x >> 5, lane = threadIdx.x & 31;
    const int M = lane & 15;
    const int hi = lane >> 4;
    const int cb = hi * 8;

    const v8f vzero = {0.f, 0.f, 0.f, 0.f, 0.f, 0.f, 0.f, 0.f};
    v8f acc[2][2];
    acc[0][0] = vzero; acc[0][1] = vzero; acc[1][0] = vzero; acc[1][1] = vzero;

    for (int i = 0; i < 5; ++i) {
        AFrag bf0, bf1;
        {
            const _Float16* p0 = wfrag + ((size_t)((0 * 5 + i) * 32 + lane) << 4);
            const _Float16* p1 = wfrag + ((size_t)((1 * 5 + i) * 32 + lane) << 4);
            bf0.h[0] = *(const v8h*)p0;  bf0.h[1] = *(const v8h*)(p0 + 8);
            bf1.h[0] = *(const v8h*)p1;  bf1.h[1] = *(const v8h*)(p1 + 8);
        }
        const int tap0 = 2 * i;
        const int tap1 = (2 * i + 1 > 8) ? 8 : 2 * i + 1;   // tap9 pad: B side is zero
        const int dt0 = tap0 / 3, df0 = tap0 % 3;
        const int dt1 = tap1 / 3, df1 = tap1 % 3;

        for (int ft = 0; ft < 2; ++ft) {
            const int f0 = (wave * 2 + ft) * 16;
            AFrag af;
            af.h[0] = *(const v8h*)&xlds[((hi ? dt1 : dt0) * W + f0 + M + (hi ? df1 : df0) * dil) * 16];
            af.h[1] = *(const v8h*)&xlds[((hi ? dt1 : dt0) * W + f0 + M + (hi ? df1 : df0) * dil) * 16 + 8];
            acc[ft][0] = __builtin_amdgcn_wmma_f32_16x16x32_f16(
                false, af.v, false, bf0.v, (short)0, acc[ft][0], false, false);
            acc[ft][1] = __builtin_amdgcn_wmma_f32_16x16x32_f16(
                false, af.v, false, bf1.v, (short)0, acc[ft][1], false, false);
        }
    }

    const int ch = lane & 15;
    const float sca = scsh[ch],      scg = scsh[16 + ch];
    const float sha = scsh[32 + ch], shg = scsh[48 + ch];
    for (int ft = 0; ft < 2; ++ft) {
        const int f0 = (wave * 2 + ft) * 16 + hi * 8;  // M = r + 8*hi
        v8h outv;
        #pragma unroll
        for (int r = 0; r < 8; ++r) {
            float a = acc[ft][0][r] * sca + sha;
            float g = acc[ft][1][r] * scg + shg;
            outv[r] = (_Float16)(tanh_fast(a) * sigf(g));
        }
        *(v8h*)&dst[(((size_t)b * 16 + ch) * TT + t) * FF + f0] = outv;
    }
}

// ---------------------------------------------------------------------------
// Grouped GRU, hidden=2, one thread per (g,b,f) sequence. Gate order r,z,n.
// ---------------------------------------------------------------------------
__global__ void gru_kernel(const _Float16* __restrict__ h4, const float* __restrict__ wih,
                           const float* __restrict__ whh, const float* __restrict__ bih,
                           const float* __restrict__ bhh, _Float16* __restrict__ merged) {
    const int tid = blockIdx.x * blockDim.x + threadIdx.x;  // G*B*F = 16384
    const int g = tid / (BB * FF);
    const int rem = tid % (BB * FF);
    const int b = rem / FF, f = rem % FF;

    float Wi[6][2], Wh[6][2], Bi[6], Bh[6];
    #pragma unroll
    for (int r = 0; r < 6; ++r) {
        Wi[r][0] = wih[(g * 6 + r) * 2 + 0];  Wi[r][1] = wih[(g * 6 + r) * 2 + 1];
        Wh[r][0] = whh[(g * 6 + r) * 2 + 0];  Wh[r][1] = whh[(g * 6 + r) * 2 + 1];
        Bi[r] = bih[g * 6 + r];               Bh[r] = bhh[g * 6 + r];
    }
    const size_t in0 = (((size_t)b * 16 + g * 2) * TT) * FF + f;
    const size_t in1 = in0 + (size_t)TT * FF;
    const size_t mb0 = (((size_t)(g * 2) * BB + b) * TT) * FF + f;
    const size_t mb1 = mb0 + (size_t)BB * TT * FF;

    float h0 = 0.0f, h1 = 0.0f;
    for (int t = 0; t < TT; ++t) {
        const float x0 = (float)h4[in0 + (size_t)t * FF];
        const float x1 = (float)h4[in1 + (size_t)t * FF];
        float gi[6], gh[6];
        #pragma unroll
        for (int r = 0; r < 6; ++r) {
            gi[r] = Wi[r][0] * x0 + Wi[r][1] * x1 + Bi[r];
            gh[r] = Wh[r][0] * h0 + Wh[r][1] * h1 + Bh[r];
        }
        const float r0 = sigf(gi[0] + gh[0]);
        const float r1 = sigf(gi[1] + gh[1]);
        const float z0 = sigf(gi[2] + gh[2]);
        const float z1 = sigf(gi[3] + gh[3]);
        const float n0 = tanh_fast(gi[4] + r0 * gh[4]);
        const float n1 = tanh_fast(gi[5] + r1 * gh[5]);
        h0 = (1.0f - z0) * n0 + z0 * h0;
        h1 = (1.0f - z1) * n1 + z1 * h1;
        merged[mb0 + (size_t)t * FF] = (_Float16)h0;
        merged[mb1 + (size_t)t * FF] = (_Float16)h1;
    }
}

// ---------------------------------------------------------------------------
// 1x1 out conv: dot over 16 channels, fp32 output.
// ---------------------------------------------------------------------------
__global__ void out_kernel(const _Float16* __restrict__ merged, const float* __restrict__ ow,
                           const float* __restrict__ ob, float* __restrict__ out) {
    const int i = blockIdx.x * blockDim.x + threadIdx.x;
    if (i >= BB * TT * FF) return;
    const int b = i / (TT * FF);
    const int r = i % (TT * FF);
    float s = ob[0];
    #pragma unroll
    for (int c = 0; c < 16; ++c)
        s += (float)merged[((size_t)c * BB + b) * TT * FF + r] * ow[c];
    out[i] = s;
}

extern "C" void kernel_launch(void* const* d_in, const int* in_sizes, int n_in,
                              void* d_out, int out_size, void* d_ws, size_t ws_size,
                              hipStream_t stream) {
    (void)in_sizes; (void)n_in; (void)out_size; (void)ws_size;
    const float* x = (const float*)d_in[0];
    const float *gw[4], *gb[4], *gg[4], *gbt[4], *gm[4], *gv[4];
    for (int g = 0; g < 4; ++g) {
        const int base = 1 + g * 6;
        gw[g]  = (const float*)d_in[base + 0];
        gb[g]  = (const float*)d_in[base + 1];
        gg[g]  = (const float*)d_in[base + 2];
        gbt[g] = (const float*)d_in[base + 3];
        gm[g]  = (const float*)d_in[base + 4];
        gv[g]  = (const float*)d_in[base + 5];
    }
    const float* gru_wih = (const float*)d_in[25];
    const float* gru_whh = (const float*)d_in[26];
    const float* gru_bih = (const float*)d_in[27];
    const float* gru_bhh = (const float*)d_in[28];
    const float* out_w   = (const float*)d_in[29];
    const float* out_b   = (const float*)d_in[30];

    char* ws = (char*)d_ws;
    _Float16* bufA = (_Float16*)ws;                 // 32 MB
    _Float16* bufB = bufA + HT;                     // 32 MB
    char* pp = ws + 2 * HT * sizeof(_Float16);
    _Float16* wfrag = (_Float16*)pp;                // 4 * 5120 halves
    float* scsh = (float*)(pp + 4 * 5120 * sizeof(_Float16));  // 4 * 64 floats

    prep1_kernel<<<1, 64, 0, stream>>>(gw[0], gb[0], gg[0], gbt[0], gm[0], gv[0],
                                       wfrag, scsh);
    for (int g = 1; g < 4; ++g)
        prep_kernel<<<1, 320, 0, stream>>>(gw[g], gb[g], gg[g], gbt[g], gm[g], gv[g],
                                           16, wfrag + g * 5120, scsh + g * 64);

    dim3 grid(TT, BB);
    gate_conv1<<<grid, 256, 0, stream>>>(x, wfrag, scsh, bufA);
    gate_conv<<<grid, 256, 0, stream>>>(bufA, wfrag + 5120,     scsh + 64,  bufB, 2);
    gate_conv<<<grid, 256, 0, stream>>>(bufB, wfrag + 2 * 5120, scsh + 128, bufA, 4);
    gate_conv<<<grid, 256, 0, stream>>>(bufA, wfrag + 3 * 5120, scsh + 192, bufB, 8);

    gru_kernel<<<64, 256, 0, stream>>>(bufB, gru_wih, gru_whh, gru_bih, gru_bhh, bufA);
    out_kernel<<<4096, 256, 0, stream>>>(bufA, out_w, out_b, (float*)d_out);
}
